// NoisyCell_41489384080067
// MI455X (gfx1250) — matrix operand order repaired and checked by reading, not compiled, not run on hardware
//
#include <hip/hip_runtime.h>
#include <stdint.h>

// Problem constants (from reference)
#define T_  512
#define B_  64
#define E_  256
#define H_  512
#define G3H (3 * H_)

typedef __attribute__((ext_vector_type(16))) __bf16 v16bf;
typedef __attribute__((ext_vector_type(8)))  __bf16 v8bf;
typedef __attribute__((ext_vector_type(8)))  float  v8f;

__device__ __forceinline__ __bf16 f2bf(float f) {
  unsigned u = __builtin_bit_cast(unsigned, f);
  unsigned r = (u + 0x7FFFu + ((u >> 16) & 1u)) >> 16;   // RNE
  unsigned short s = (unsigned short)r;
  return __builtin_bit_cast(__bf16, s);
}

__device__ __forceinline__ float sigm(float x) {
  return 1.0f / (1.0f + __expf(-x));
}

__device__ __forceinline__ v8f wmma_bf16(v16bf a, v16bf b, v8f c) {
  return __builtin_amdgcn_wmma_f32_16x16x32_bf16(false, a, false, b, (short)0, c,
                                                 false, false);
}

// ---------------------------------------------------------------- converts --
__global__ void k_f32_to_bf16(const float* __restrict__ s, __bf16* __restrict__ d,
                              int n) {
  int i = blockIdx.x * blockDim.x + threadIdx.x;
  if (i < n) d[i] = f2bf(s[i]);
}

__global__ void k_zero_state(float* __restrict__ h0f, float* __restrict__ h1f,
                             __bf16* __restrict__ h0b, __bf16* __restrict__ h1b) {
  int i = blockIdx.x * blockDim.x + threadIdx.x;
  if (i < B_ * H_) {
    h0f[i] = 0.0f; h1f[i] = 0.0f;
    __bf16 z = f2bf(0.0f);
    h0b[i] = z; h1b[i] = z;
  }
}

// ------------------------------------------------------- big input GEMMs ----
// C[M, 3H] = A[M, K] @ W[3H, K]^T + bias ; A,W bf16 row-major, C f32.
// grid = (M/16, 3), block = 256 (8 waves). wave w covers 64 cols.
__global__ void k_gemm_bias(const __bf16* __restrict__ A,
                            const __bf16* __restrict__ W,
                            const float* __restrict__ bias,
                            float* __restrict__ C, int K) {
  const int lane = threadIdx.x & 31;
  const int wave = threadIdx.x >> 5;
  const int g    = lane >> 4;      // lane half-group
  const int lr   = lane & 15;
  const int mBase = blockIdx.x * 16;
  const int nBase = blockIdx.y * 512 + wave * 64;

  const __bf16* arow = A + (size_t)(mBase + lr) * K + g * 16;

  v8f acc[4];
#pragma unroll
  for (int i = 0; i < 4; ++i) {
    float b = bias[nBase + i * 16 + lr];
#pragma unroll
    for (int v = 0; v < 8; ++v) acc[i][v] = b;
  }

  for (int k = 0; k < K; k += 32) {
    v8bf alo = *(const v8bf*)(arow + k);
    v8bf ahi = *(const v8bf*)(arow + k + 8);
    v16bf af = __builtin_shufflevector(alo, ahi, 0, 1, 2, 3, 4, 5, 6, 7, 8, 9,
                                       10, 11, 12, 13, 14, 15);
#pragma unroll
    for (int i = 0; i < 4; ++i) {
      const __bf16* brow = W + (size_t)(nBase + i * 16 + lr) * K + k + g * 16;
      v16bf bf = *(const v16bf*)brow;
      acc[i] = wmma_bf16(af, bf, acc[i]);
    }
  }

#pragma unroll
  for (int i = 0; i < 4; ++i) {
    const int col = nBase + i * 16 + lr;
#pragma unroll
    for (int v = 0; v < 8; ++v)
      C[(size_t)(mBase + v + 8 * g) * G3H + col] = acc[i][v];
  }
}

// ------------------------------------------------------------ GRU step ------
// One timestep of one layer: gh = h_in @ Whh^T, gates, noise, length mask.
// grid = 16 blocks x 256 threads = 128 waves; wave -> one 16x16 tile of h_new.
__global__ void k_gru_step(const __bf16* __restrict__ h_in_bf,  // [B,H]
                           const float*  __restrict__ h_in_f,   // [B,H]
                           __bf16* __restrict__ h_out_bf,
                           float*  __restrict__ h_out_f,
                           __bf16* __restrict__ h_all_bf,       // t-slice or null
                           const float* __restrict__ gi_t,      // [B,3H]
                           const __bf16* __restrict__ Whh,      // [3H,H]
                           const float* __restrict__ bhh,       // [3H]
                           const float* __restrict__ noise_t,   // [B,H]
                           const int* __restrict__ lengths, int t) {
  const int lane = threadIdx.x & 31;
  const int wave = threadIdx.x >> 5;
  const int w    = blockIdx.x * 8 + wave;
  const int g    = lane >> 4;
  const int lr   = lane & 15;
  const int mBase   = (w & 3) * 16;        // 4 row tiles
  const int colBase = (w >> 2) * 16;       // 32 col tiles
  const int col  = colBase + lr;

  v8f accR, accZ, accN, giN;
#pragma unroll
  for (int v = 0; v < 8; ++v) {
    int m = mBase + v + 8 * g;
    const float* gr = gi_t + (size_t)m * G3H;
    accR[v] = gr[col]           + bhh[col];            // fold i_r + b_hh_r
    accZ[v] = gr[H_ + col]      + bhh[H_ + col];       // fold i_z + b_hh_z
    accN[v] = bhh[2 * H_ + col];                       // h_n gets b_hh_n only
    giN[v]  = gr[2 * H_ + col];                        // i_n kept separate
  }

  const __bf16* arow = h_in_bf + (size_t)(mBase + lr) * H_ + g * 16;
  const __bf16* br0  = Whh + (size_t)(col) * H_ + g * 16;
  const __bf16* br1  = Whh + (size_t)(H_ + col) * H_ + g * 16;
  const __bf16* br2  = Whh + (size_t)(2 * H_ + col) * H_ + g * 16;

  for (int k = 0; k < H_; k += 32) {
    v8bf alo = *(const v8bf*)(arow + k);
    v8bf ahi = *(const v8bf*)(arow + k + 8);
    v16bf af = __builtin_shufflevector(alo, ahi, 0, 1, 2, 3, 4, 5, 6, 7, 8, 9,
                                       10, 11, 12, 13, 14, 15);
    v16bf b0 = *(const v16bf*)(br0 + k);
    v16bf b1 = *(const v16bf*)(br1 + k);
    v16bf b2 = *(const v16bf*)(br2 + k);
    accR = wmma_bf16(af, b0, accR);
    accZ = wmma_bf16(af, b1, accZ);
    accN = wmma_bf16(af, b2, accN);
  }

#pragma unroll
  for (int v = 0; v < 8; ++v) {
    int m = mBase + v + 8 * g;
    size_t idx = (size_t)m * H_ + col;
    float r  = sigm(accR[v]);
    float z  = sigm(accZ[v]);
    float nn = tanhf(giN[v] + r * accN[v]);
    float hprev = h_in_f[idx];
    float hnew  = (1.0f - z) * nn + z * hprev + 0.1f * noise_t[idx];
    bool active = t < lengths[m];
    float hf = active ? hnew : hprev;
    h_out_f[idx] = hf;
    __bf16 hb = f2bf(hf);
    h_out_bf[idx] = hb;
    if (h_all_bf) h_all_bf[idx] = hb;   // feeds layer-1 input GEMM
  }
}

// ------------------------------------------------------------ finalize ------
__global__ void k_finalize(const float* __restrict__ h0f,
                           const float* __restrict__ h1f,
                           const int* __restrict__ unsorted,
                           float* __restrict__ out) {
  int i = blockIdx.x * blockDim.x + threadIdx.x;
  if (i < 2 * B_ * H_) {
    int l   = i / (B_ * H_);
    int rem = i - l * (B_ * H_);
    int b   = rem / H_;
    int c   = rem - b * H_;
    const float* src = (l == 0) ? h0f : h1f;
    out[i] = src[(size_t)unsorted[b] * H_ + c];
  }
}

// ------------------------------------------------------------ launcher ------
extern "C" void kernel_launch(void* const* d_in, const int* in_sizes, int n_in,
                              void* d_out, int out_size, void* d_ws, size_t ws_size,
                              hipStream_t stream) {
  (void)in_sizes; (void)n_in; (void)out_size; (void)ws_size;
  const float* x      = (const float*)d_in[0];
  const float* noise  = (const float*)d_in[1];
  const float* W_ih0  = (const float*)d_in[2];
  const float* W_hh0  = (const float*)d_in[3];
  const float* b_ih0  = (const float*)d_in[4];
  const float* b_hh0  = (const float*)d_in[5];
  const float* W_ih1  = (const float*)d_in[6];
  const float* W_hh1  = (const float*)d_in[7];
  const float* b_ih1  = (const float*)d_in[8];
  const float* b_hh1  = (const float*)d_in[9];
  const int* lengths  = (const int*)d_in[10];
  const int* unsorted = (const int*)d_in[11];
  float* out = (float*)d_out;

  char* ws = (char*)d_ws;
  size_t off = 0;
  auto take = [&](size_t bytes) -> char* {
    char* p = ws + off;
    off = (off + bytes + 255) & ~(size_t)255;
    return p;
  };

  __bf16* wih0b = (__bf16*)take((size_t)G3H * E_ * 2);
  __bf16* whh0b = (__bf16*)take((size_t)G3H * H_ * 2);
  __bf16* wih1b = (__bf16*)take((size_t)G3H * H_ * 2);
  __bf16* whh1b = (__bf16*)take((size_t)G3H * H_ * 2);
  __bf16* xb    = (__bf16*)take((size_t)T_ * B_ * E_ * 2);
  __bf16* h0all = (__bf16*)take((size_t)T_ * B_ * H_ * 2);
  float*  gi0   = (float*)take((size_t)T_ * B_ * G3H * 4);
  float*  gi1   = (float*)take((size_t)T_ * B_ * G3H * 4);
  float*  h0f[2] = {(float*)take(B_ * H_ * 4), (float*)take(B_ * H_ * 4)};
  float*  h1f[2] = {(float*)take(B_ * H_ * 4), (float*)take(B_ * H_ * 4)};
  __bf16* h0b[2] = {(__bf16*)take(B_ * H_ * 2), (__bf16*)take(B_ * H_ * 2)};
  __bf16* h1b[2] = {(__bf16*)take(B_ * H_ * 2), (__bf16*)take(B_ * H_ * 2)};

  auto cvt = [&](const float* s, __bf16* d, int n) {
    k_f32_to_bf16<<<(n + 255) / 256, 256, 0, stream>>>(s, d, n);
  };
  cvt(W_ih0, wih0b, G3H * E_);
  cvt(W_hh0, whh0b, G3H * H_);
  cvt(W_ih1, wih1b, G3H * H_);
  cvt(W_hh1, whh1b, G3H * H_);
  cvt(x, xb, T_ * B_ * E_);
  k_zero_state<<<(B_ * H_ + 255) / 256, 256, 0, stream>>>(h0f[0], h1f[0],
                                                          h0b[0], h1b[0]);

  dim3 ggrid(T_ * B_ / 16, 3);

  // Phase A: gi0 = x @ W_ih0^T + b_ih0   (time-parallel WMMA GEMM)
  k_gemm_bias<<<ggrid, 256, 0, stream>>>(xb, wih0b, b_ih0, gi0, E_);

  // Phase B: layer-0 recurrence (stream-ordered step kernels, ping-pong h)
  int cur = 0;
  for (int t = 0; t < T_; ++t) {
    int nxt = cur ^ 1;
    k_gru_step<<<16, 256, 0, stream>>>(
        h0b[cur], h0f[cur], h0b[nxt], h0f[nxt],
        h0all + (size_t)t * B_ * H_,
        gi0 + (size_t)t * B_ * G3H,
        whh0b, b_hh0,
        noise + ((size_t)t * 2 + 0) * B_ * H_,
        lengths, t);
    cur = nxt;
  }
  const int h0fin = cur;  // ends at 0 (512 flips)

  // Phase C: gi1 = h0_all @ W_ih1^T + b_ih1  (time-parallel WMMA GEMM)
  k_gemm_bias<<<ggrid, 256, 0, stream>>>(h0all, wih1b, b_ih1, gi1, H_);

  // Phase D: layer-1 recurrence
  cur = 0;
  for (int t = 0; t < T_; ++t) {
    int nxt = cur ^ 1;
    k_gru_step<<<16, 256, 0, stream>>>(
        h1b[cur], h1f[cur], h1b[nxt], h1f[nxt],
        (__bf16*)nullptr,
        gi1 + (size_t)t * B_ * G3H,
        whh1b, b_hh1,
        noise + ((size_t)t * 2 + 1) * B_ * H_,
        lengths, t);
    cur = nxt;
  }

  k_finalize<<<(2 * B_ * H_ + 255) / 256, 256, 0, stream>>>(h0f[h0fin], h1f[cur],
                                                            unsorted, out);
}
